// EdgeScorePredictor_25812753449665
// MI455X (gfx1250) — matrix operand order repaired and checked by reading, not compiled, not run on hardware
//
#include <hip/hip_runtime.h>
#include <cstdint>
#include <cstddef>

// ---------------- CDNA5 (gfx1250) wave32 WMMA types ----------------
typedef __attribute__((ext_vector_type(16))) __bf16 v16bf;
typedef __attribute__((ext_vector_type(8)))  float  v8f;

#define WAVES            8
#define MSUB             2                    // 16-edge tiles per compute pass
#define PASSES_PER_WAVE  2
#define EDGES_PER_WAVE   (MSUB * PASSES_PER_WAVE * 16)       // 64
#define EDGES_PER_BLOCK  (WAVES * EDGES_PER_WAVE)            // 512

// Fragment-major LDS layout: each 16x32 bf16 fragment = 1024B block:
//   bytes [   0..511]: q0 of lanes 0..31 (16B each, lane-contiguous)
//   bytes [512..1023]: q1 of lanes 0..31
// => every ds_load_b128 is lane-consecutive 16B: conflict-free across 64 banks.
#define FRAG     1024
#define OFF_W1   0                            // 64 frags (kc*8+n)
#define OFF_W2   (OFF_W1 + 64*FRAG)           // 65536: 32 frags (kc*8+n)
#define OFF_W3   (OFF_W2 + 32*FRAG)           // 98304: 4 frags (kc), cols 1..15 zero
#define OFF_B1   (OFF_W3 + 4*FRAG)            // 102400
#define OFF_B2   (OFF_B1 + 128*4)             // 102912
#define OFF_B3   (OFF_B2 + 128*4)             // 103424
#define OFF_EF   (OFF_B3 + 16)                // 103440: per wave MSUB*8 frags = 16384
#define OFF_H1   (OFF_EF + WAVES*MSUB*8*FRAG) // 234512: per wave MSUB*4 frags = 8192
#define SMEM_BYTES (OFF_H1 + WAVES*MSUB*4*FRAG) // 300048 (< 320KB WGP LDS)

union Frag {
    v16bf v;
    uint4 q[2];
};

__device__ __forceinline__ uint32_t f2bf(float f) {
    uint32_t u = __float_as_uint(f);
    return (u + 0x7FFFu + ((u >> 16) & 1u)) >> 16;   // round-to-nearest-even
}
__device__ __forceinline__ uint32_t pack2bf(float lo, float hi) {
    return f2bf(lo) | (f2bf(hi) << 16);
}

// Load one 16x32 bf16 fragment from fragment-major LDS block.
__device__ __forceinline__ v16bf ldFrag(const char* fragBase, int lane) {
    Frag f;
    f.q[0] = *(const uint4*)(fragBase + lane * 16);
    f.q[1] = *(const uint4*)(fragBase + 512 + lane * 16);
    return f.v;
}

__device__ __forceinline__ v8f wmma_bf16(v16bf a, v16bf b, v8f c) {
    return __builtin_amdgcn_wmma_f32_16x16x32_bf16(false, a, false, b, (short)0, c, false, false);
}

// Scatter one packed pair of bf16 (8B) for columns cf..cf+3 of row `row`
// into a fragment-major A staging area (tileStride bytes per 16-row tile).
__device__ __forceinline__ void stScatter8(char* base, int row, int cf,
                                           uint32_t lo, uint32_t hi, int tileStride) {
    const int tile  = row >> 4;
    const int r16   = row & 15;
    const int kc    = cf >> 5;
    const int lane2 = r16 + (((cf >> 3) & 1) << 4);
    const int sh    = (cf >> 4) & 1;       // which 512B half
    const int sl    = cf & 7;              // 0 or 4
    *(uint2*)(base + tile * tileStride + kc * FRAG + sh * 512 + lane2 * 16 + sl * 2)
        = make_uint2(lo, hi);
}

__global__ void __launch_bounds__(256)
edge_mlp_wmma_kernel(const float* __restrict__ node_rep,
                     const long long* __restrict__ eidx,   // [2, E] int64
                     const float* __restrict__ W1, const float* __restrict__ b1,
                     const float* __restrict__ W2, const float* __restrict__ b2,
                     const float* __restrict__ W3, const float* __restrict__ b3,
                     float* __restrict__ out, int E)
{
    extern __shared__ char smem[];
    const int tid  = threadIdx.x;
    const int lane = tid & 31;
    const int wave = tid >> 5;

    // ---------- Stage weights (f32 -> bf16) into fragment-major LDS ----------
    {
        // W1: [256][128] f32 = 8192 float4 (k = i>>5, c = (i&31)*4)
        const float4* w1v = (const float4*)W1;
        #pragma unroll 4
        for (int i = tid; i < 8192; i += 256) {
            float4 f = w1v[i];
            const int k = i >> 5, c = (i & 31) * 4;
            const int kc = k >> 5, kk = k & 31, n = c >> 4;
            const int sh = (kk >> 4) & 1, sl = kk & 7, ln = (c & 15) + (((kk >> 3) & 1) << 4);
            char* p = smem + OFF_W1 + (kc * 8 + n) * FRAG + sh * 512 + ln * 16 + sl * 2;
            *(uint16_t*)(p) = (uint16_t)f2bf(f.x); *(uint16_t*)(p + 16) = (uint16_t)f2bf(f.y);
            *(uint16_t*)(p + 32) = (uint16_t)f2bf(f.z); *(uint16_t*)(p + 48) = (uint16_t)f2bf(f.w);
        }
        // W2: [128][128] f32 = 4096 float4
        const float4* w2v = (const float4*)W2;
        #pragma unroll 4
        for (int i = tid; i < 4096; i += 256) {
            float4 f = w2v[i];
            const int k = i >> 5, c = (i & 31) * 4;
            const int kc = k >> 5, kk = k & 31, n = c >> 4;
            const int sh = (kk >> 4) & 1, sl = kk & 7, ln = (c & 15) + (((kk >> 3) & 1) << 4);
            char* p = smem + OFF_W2 + (kc * 8 + n) * FRAG + sh * 512 + ln * 16 + sl * 2;
            *(uint16_t*)(p) = (uint16_t)f2bf(f.x); *(uint16_t*)(p + 16) = (uint16_t)f2bf(f.y);
            *(uint16_t*)(p + 32) = (uint16_t)f2bf(f.z); *(uint16_t*)(p + 48) = (uint16_t)f2bf(f.w);
        }
        // W3 zero-padded frags [4][1024]
        *(uint4*)(smem + OFF_W3 + tid * 16) = make_uint4(0u, 0u, 0u, 0u);
        if (tid < 128) {
            ((float*)(smem + OFF_B1))[tid] = b1[tid];
            ((float*)(smem + OFF_B2))[tid] = b2[tid];
        }
        if (tid == 0) *(float*)(smem + OFF_B3) = b3[0];
    }
    __syncthreads();
    if (tid < 128) {   // W3 col 0: k = tid -> frag k>>5, lane 16*bit3(k), half bit4(k), slot k&7
        *(uint16_t*)(smem + OFF_W3 + (tid >> 5) * FRAG + ((tid >> 4) & 1) * 512 + ((tid >> 3) & 1) * 256 + (tid & 7) * 2) = (uint16_t)f2bf(W3[tid]);
    }
    __syncthreads();

    char* efW = smem + OFF_EF + wave * (MSUB * 8 * FRAG);   // 2 tiles x 8 A-frags
    char* h1W = smem + OFF_H1 + wave * (MSUB * 4 * FRAG);   // 2 tiles x 4 A-frags
    const char* w1f = smem + OFF_W1;
    const char* w2f = smem + OFF_W2;
    const char* w3f = smem + OFF_W3;
    const float* b1s = (const float*)(smem + OFF_B1);
    const float* b2s = (const float*)(smem + OFF_B2);
    const float  b3v = *(const float*)(smem + OFF_B3);

    const int nc = lane & 15;          // C/D layout: N column
    const int kh = lane >> 4;          // C/D layout: row-half select
    const int hiHalf = (nc >= 8) ? 16 : 0;   // lane2 adjust for h writeback

    for (int it = 0; it < PASSES_PER_WAVE; ++it) {
        const int eBase = blockIdx.x * EDGES_PER_BLOCK + wave * EDGES_PER_WAVE + it * (MSUB * 16);

        // ---------- Gather + convert: 32 edges x 256 bf16, into A-frag layout ----------
        // 32 rows * 64 float4 = 2048 float4; 32 lanes -> 64 iters; each iter is one
        // contiguous 512B half-row from node_rep (coalesced, L2-resident).
        #pragma unroll 4
        for (int j = 0; j < 64; ++j) {
            const int idx  = j * 32 + lane;
            const int row  = idx >> 6;          // 0..31
            const int q    = idx & 63;          // float4 index within 256-col row
            const int half = q >> 5;            // 0 = src, 1 = dst
            long long e = (long long)(eBase + row);
            if (e >= E) e = E - 1;              // clamp tail (store masked later)
            const long long node = eidx[(long long)half * E + e];
            const float4 f = *(const float4*)(node_rep + node * 128 + (q & 31) * 4);
            stScatter8(efW, row, q * 4, pack2bf(f.x, f.y), pack2bf(f.z, f.w), 8 * FRAG);
        }

        // ---------- Layer 1: [32x256] x [256x128], B-frag shared by 2 M-tiles ----------
        v8f acc[MSUB][8];
        #pragma unroll
        for (int t = 0; t < MSUB; ++t)
            #pragma unroll
            for (int n = 0; n < 8; ++n) acc[t][n] = v8f{};
        #pragma unroll
        for (int kc = 0; kc < 8; ++kc) {
            const v16bf a0 = ldFrag(efW + kc * FRAG, lane);
            const v16bf a1 = ldFrag(efW + 8 * FRAG + kc * FRAG, lane);
            #pragma unroll
            for (int n = 0; n < 8; ++n) {
                const v16bf b = ldFrag(w1f + (kc * 8 + n) * FRAG, lane);
                acc[0][n] = wmma_bf16(a0, b, acc[0][n]);
                acc[1][n] = wmma_bf16(a1, b, acc[1][n]);
            }
        }
        // bias + ReLU -> h1 in fragment-major A layout
        #pragma unroll
        for (int t = 0; t < MSUB; ++t) {
            char* h1t = h1W + t * (4 * FRAG);
            #pragma unroll
            for (int n = 0; n < 8; ++n) {
                const float bias = b1s[n * 16 + nc];
                char* fb = h1t + (n >> 1) * FRAG + (n & 1) * 512 + (nc & 7) * 2;
                #pragma unroll
                for (int j = 0; j < 8; ++j) {
                    float v = acc[t][n][j] + bias;
                    v = v > 0.f ? v : 0.f;
                    *(uint16_t*)(fb + (j + 8 * kh + hiHalf) * 16) = (uint16_t)f2bf(v);
                }
            }
        }

        // ---------- Layer 2: [32x128] x [128x128] ----------
        v8f acc2[MSUB][8];
        #pragma unroll
        for (int t = 0; t < MSUB; ++t)
            #pragma unroll
            for (int n = 0; n < 8; ++n) acc2[t][n] = v8f{};
        #pragma unroll
        for (int kc = 0; kc < 4; ++kc) {
            const v16bf a0 = ldFrag(h1W + kc * FRAG, lane);
            const v16bf a1 = ldFrag(h1W + 4 * FRAG + kc * FRAG, lane);
            #pragma unroll
            for (int n = 0; n < 8; ++n) {
                const v16bf b = ldFrag(w2f + (kc * 8 + n) * FRAG, lane);
                acc2[0][n] = wmma_bf16(a0, b, acc2[0][n]);
                acc2[1][n] = wmma_bf16(a1, b, acc2[1][n]);
            }
        }
        // bias + ReLU -> h2, reuse ef region (frags 0..3 of each tile)
        #pragma unroll
        for (int t = 0; t < MSUB; ++t) {
            char* h2t = efW + t * (8 * FRAG);
            #pragma unroll
            for (int n = 0; n < 8; ++n) {
                const float bias = b2s[n * 16 + nc];
                char* fb = h2t + (n >> 1) * FRAG + (n & 1) * 512 + (nc & 7) * 2;
                #pragma unroll
                for (int j = 0; j < 8; ++j) {
                    float v = acc2[t][n][j] + bias;
                    v = v > 0.f ? v : 0.f;
                    *(uint16_t*)(fb + (j + 8 * kh + hiHalf) * 16) = (uint16_t)f2bf(v);
                }
            }
        }

        // ---------- Layer 3: [32x128] x [128x16] (only col 0 valid) ----------
        v8f acc3[MSUB];
        #pragma unroll
        for (int t = 0; t < MSUB; ++t) acc3[t] = v8f{};
        #pragma unroll
        for (int kc = 0; kc < 4; ++kc) {
            const v16bf a0 = ldFrag(efW + kc * FRAG, lane);
            const v16bf a1 = ldFrag(efW + 8 * FRAG + kc * FRAG, lane);
            const v16bf b  = ldFrag(w3f + kc * FRAG, lane);
            acc3[0] = wmma_bf16(a0, b, acc3[0]);
            acc3[1] = wmma_bf16(a1, b, acc3[1]);
        }

        // sigmoid + store: lanes with nc==0 hold column 0 (the logit)
        if (nc == 0) {
            #pragma unroll
            for (int t = 0; t < MSUB; ++t) {
                #pragma unroll
                for (int j = 0; j < 8; ++j) {
                    const int e = eBase + t * 16 + j + 8 * kh;
                    if (e < E) {
                        const float z = acc3[t][j] + b3v;
                        out[e] = 1.0f / (1.0f + __expf(-z));
                    }
                }
            }
        }
    }
}

extern "C" void kernel_launch(void* const* d_in, const int* in_sizes, int n_in,
                              void* d_out, int out_size, void* d_ws, size_t ws_size,
                              hipStream_t stream) {
    const float*     node_rep = (const float*)d_in[0];
    const long long* eidx     = (const long long*)d_in[1];
    const float*     W1 = (const float*)d_in[2];
    const float*     b1 = (const float*)d_in[3];
    const float*     W2 = (const float*)d_in[4];
    const float*     b2 = (const float*)d_in[5];
    const float*     W3 = (const float*)d_in[6];
    const float*     b3 = (const float*)d_in[7];
    float*           out = (float*)d_out;

    const int E = in_sizes[1] / 2;   // edge_index is [2, E]

    (void)hipFuncSetAttribute((const void*)edge_mlp_wmma_kernel,
                              hipFuncAttributeMaxDynamicSharedMemorySize, SMEM_BYTES);

    const int grid = (E + EDGES_PER_BLOCK - 1) / EDGES_PER_BLOCK;
    edge_mlp_wmma_kernel<<<grid, 256, SMEM_BYTES, stream>>>(
        node_rep, eidx, W1, b1, W2, b2, W3, b3, out, E);
}
